// GruMaskedDecoder_15436112462545
// MI455X (gfx1250) — matrix operand order repaired and checked by reading, not compiled, not run on hardware
//
#include <hip/hip_runtime.h>

// ---------------------------------------------------------------------------
// GRU masked decoder for MI455X (gfx1250).
// B=64, S=512, I=512, H=1024.
// Strategy:
//  - bf16 WMMA (v_wmma_f32_16x16x32_bf16) for all matmuls, f32 accumulate.
//  - Fuse x-projection into the recurrent GEMM: K = H + I = 1536 against
//    Wcat = [W_hh | W_ih].  r/z gates fuse fully; n gate keeps xn/hn split.
//  - One kernel per time step (512 graph nodes give serial ordering).
//  - h_drop (dropout+mask applied, bf16) is produced by the previous step's
//    epilogue into a ping-pong workspace buffer; hidden state itself stays f32
//    in d_out so rounding does not compound across steps.
// ---------------------------------------------------------------------------

#define B_  64
#define S_  512
#define I_  512
#define H_  1024
#define KC_ 1536   // H + I

typedef __attribute__((ext_vector_type(16))) __bf16 v16bf;
typedef __attribute__((ext_vector_type(8)))  float  v8f;

struct alignas(16) U128 { unsigned int x[4]; };
union Frag { v16bf v; U128 q[2]; };

struct alignas(16) F4  { float x, y, z, w; };
struct alignas(8)  US4 { unsigned short x, y, z, w; };

__device__ __forceinline__ unsigned short f2bf(float f) {
  unsigned int u = __float_as_uint(f);
  unsigned int r = u + 0x7FFFu + ((u >> 16) & 1u);   // round-to-nearest-even
  return (unsigned short)(r >> 16);
}

__device__ __forceinline__ float sigmoid_(float x) {
  return 1.0f / (1.0f + __expf(-x));
}
__device__ __forceinline__ float tanh_(float x) {
  x = fminf(8.0f, fmaxf(-8.0f, x));
  float e = __expf(2.0f * x);
  return (e - 1.0f) / (e + 1.0f);
}

__device__ __forceinline__ v16bf load_frag(const unsigned short* p0,
                                           const unsigned short* p1) {
  Frag f;
  f.q[0] = *(const U128*)p0;
  f.q[1] = *(const U128*)p1;
  return f.v;
}

__device__ __forceinline__ v8f wmma_bf16(v16bf a, v16bf b, v8f c) {
  return __builtin_amdgcn_wmma_f32_16x16x32_bf16(
      /*neg_a=*/false, a, /*neg_b=*/false, b,
      /*c_mod=*/(short)0, c, /*reuse_a=*/false, /*reuse_b=*/false);
}

// ---------------------------------------------------------------------------
// Prep kernels
// ---------------------------------------------------------------------------
__global__ __launch_bounds__(256) void cvt_x_bf16(const F4* __restrict__ in,
                                                  US4* __restrict__ out, int n4) {
  int i = blockIdx.x * blockDim.x + threadIdx.x;
  if (i < n4) {
    F4 f = in[i];
    US4 u;
    u.x = f2bf(f.x); u.y = f2bf(f.y); u.z = f2bf(f.z); u.w = f2bf(f.w);
    out[i] = u;
  }
}

// Wcat[g, k] : k < H -> W_hh[g, k]; else W_ih[g, k-H].  Row-major, stride KC_.
__global__ __launch_bounds__(256) void build_wcat(const float* __restrict__ whh,
                                                  const float* __restrict__ wih,
                                                  unsigned short* __restrict__ wcat,
                                                  int n4) {
  int i = blockIdx.x * blockDim.x + threadIdx.x;
  if (i < n4) {
    int base = i * 4;
    int g = base / KC_;
    int k = base % KC_;
    const float* src = (k < H_) ? (whh + (size_t)g * H_ + k)
                                : (wih + (size_t)g * I_ + (k - H_));
    F4 f = *(const F4*)src;
    US4 u;
    u.x = f2bf(f.x); u.y = f2bf(f.y); u.z = f2bf(f.z); u.w = f2bf(f.w);
    *(US4*)(wcat + base) = u;
  }
}

// hd_bf[b,k] = init_hidden * (dropout_mask*m + (1-m)) with m = mask[b,0]
__global__ __launch_bounds__(256) void init_hdrop(const float* __restrict__ h0,
                                                  const float* __restrict__ dmask,
                                                  const float* __restrict__ mask,
                                                  unsigned short* __restrict__ hd) {
  int t = blockIdx.x * blockDim.x + threadIdx.x;
  if (t < B_ * H_) {
    int b = t >> 10;
    float m = mask[(size_t)b * S_];
    float v = h0[t] * (dmask[t] * m + 1.0f - m);
    hd[t] = f2bf(v);
  }
}

// ---------------------------------------------------------------------------
// One GRU time step.  grid = 16 blocks (64-col tile of H), block = 512 thr.
// Wave (wm, wn): M rows [16*wm, 16*wm+16), N cols jt*64 + 16*wn + [0,16).
// ---------------------------------------------------------------------------
__global__ __launch_bounds__(512)
void gru_step(const unsigned short* __restrict__ xbf,
              const unsigned short* __restrict__ wcat,
              const unsigned short* __restrict__ hdcur,
              unsigned short* __restrict__ hdnext,
              const float* __restrict__ b_ih,
              const float* __restrict__ b_hh,
              const float* __restrict__ dmask,
              const float* __restrict__ mask,
              const float* __restrict__ hprev, long hstride,
              float* __restrict__ out, int s)
{
  const int tid  = threadIdx.x;
  const int lane = tid & 31;
  const int w    = tid >> 5;
  const int wm   = w & 3;          // 4 waves over M (batch)
  const int wn   = w >> 2;         // 4 waves over N (16 cols each)
  const int jt   = blockIdx.x;     // 64-col tile of H
  const int ln   = lane & 15;
  const int kh   = lane >> 4;      // lane half (K packing)

  const int mrow = wm * 16 + ln;                 // batch row for A loads
  const int col  = jt * 64 + wn * 16 + ln;       // hidden column / B column

  v8f acc_r  = {0.f,0.f,0.f,0.f,0.f,0.f,0.f,0.f};
  v8f acc_z  = {0.f,0.f,0.f,0.f,0.f,0.f,0.f,0.f};
  v8f acc_hn = {0.f,0.f,0.f,0.f,0.f,0.f,0.f,0.f};
  v8f acc_xn = {0.f,0.f,0.f,0.f,0.f,0.f,0.f,0.f};

  // A: this code's K order (same for B; not the ISA table) -> lane half kh holds contiguous K = kh*16 + [0,16)
  const unsigned short* arow = hdcur + (size_t)mrow * H_;
  // B: this code's K order (same as A; not the ISA table) -> lane half kh holds contiguous K = kh*16 + [0,16)
  const unsigned short* br = wcat + (size_t)(0 * H_ + col) * KC_ + kh * 16;
  const unsigned short* bz = wcat + (size_t)(1 * H_ + col) * KC_ + kh * 16;
  const unsigned short* bn = wcat + (size_t)(2 * H_ + col) * KC_ + kh * 16;

  // ---- K part 1: hidden projection (K over H) ----
  #pragma unroll 4
  for (int kk = 0; kk < H_; kk += 32) {
    v16bf a  = load_frag(arow + kk + kh * 16, arow + kk + kh * 16 + 8);
    v16bf b0 = load_frag(br + kk, br + kk + 8);
    v16bf b1 = load_frag(bz + kk, bz + kk + 8);
    v16bf b2 = load_frag(bn + kk, bn + kk + 8);
    acc_r  = wmma_bf16(a, b0, acc_r);
    acc_z  = wmma_bf16(a, b1, acc_z);
    acc_hn = wmma_bf16(a, b2, acc_hn);
  }

  // ---- K part 2: input projection (K over I), fused into r/z, separate xn ----
  const unsigned short* xrow = xbf + ((size_t)mrow * S_ + s) * I_;
  #pragma unroll 4
  for (int kk = 0; kk < I_; kk += 32) {
    v16bf a  = load_frag(xrow + kk + kh * 16, xrow + kk + kh * 16 + 8);
    v16bf b0 = load_frag(br + H_ + kk, br + H_ + kk + 8);
    v16bf b1 = load_frag(bz + H_ + kk, bz + H_ + kk + 8);
    v16bf b2 = load_frag(bn + H_ + kk, bn + H_ + kk + 8);
    acc_r  = wmma_bf16(a, b0, acc_r);
    acc_z  = wmma_bf16(a, b1, acc_z);
    acc_xn = wmma_bf16(a, b2, acc_xn);
  }

  // ---- epilogue: gates + blend in f32, emit next step's bf16 h_drop ----
  const float bihr = b_ih[col],        bhhr = b_hh[col];
  const float bihz = b_ih[H_ + col],   bhhz = b_hh[H_ + col];
  const float bihn = b_ih[2*H_ + col], bhhn = b_hh[2*H_ + col];

  #pragma unroll
  for (int v = 0; v < 8; ++v) {
    // C/D layout: VGPR v, lanes 0-15 -> M=v, lanes 16-31 -> M=v+8
    const int row = wm * 16 + v + (kh << 3);
    float r = sigmoid_(acc_r[v] + bihr + bhhr);
    float z = sigmoid_(acc_z[v] + bihz + bhhz);
    float n = tanh_(acc_xn[v] + bihn + r * (acc_hn[v] + bhhn));

    float hp = hprev[(size_t)row * hstride + col];     // previous hidden (f32)
    float dm = dmask[(size_t)row * H_ + col];
    float m  = mask[(size_t)row * S_ + s];
    float hd = hp * (dm * m + 1.0f - m);               // dropped hidden (f32)

    float cur = (1.0f - z) * n + z * hd;
    float ho  = m * cur + (1.0f - m) * hd;

    out[((size_t)row * S_ + s) * H_ + col] = ho;
    if (s == S_ - 1)
      out[(size_t)B_ * S_ * H_ + (size_t)row * H_ + col] = ho;  // last_hidden

    // produce next step's dropped-hidden operand (bf16), reusing dm
    float mn  = (s + 1 < S_) ? mask[(size_t)row * S_ + s + 1] : 0.0f;
    float hdn = ho * (dm * mn + 1.0f - mn);
    hdnext[(size_t)row * H_ + col] = f2bf(hdn);
  }
}

// ---------------------------------------------------------------------------
extern "C" void kernel_launch(void* const* d_in, const int* in_sizes, int n_in,
                              void* d_out, int out_size, void* d_ws, size_t ws_size,
                              hipStream_t stream) {
  const float* x     = (const float*)d_in[0];   // [B,S,I]
  const float* mask  = (const float*)d_in[1];   // [B,S]
  const float* h0    = (const float*)d_in[2];   // [B,H]
  const float* dmask = (const float*)d_in[3];   // [B,H]
  const float* wih   = (const float*)d_in[4];   // [3H,I]
  const float* whh   = (const float*)d_in[5];   // [3H,H]
  const float* bih   = (const float*)d_in[6];   // [3H]
  const float* bhh   = (const float*)d_in[7];   // [3H]
  float* out = (float*)d_out;                   // [B,S,H] ++ [B,H]

  // workspace layout (bf16 stored as ushort)
  unsigned short* xbf  = (unsigned short*)d_ws;                 // B*S*I
  unsigned short* wcat = xbf + (size_t)B_ * S_ * I_;            // 3H*KC
  unsigned short* hdA  = wcat + (size_t)3 * H_ * KC_;           // B*H
  unsigned short* hdB  = hdA + (size_t)B_ * H_;                 // B*H

  {
    int n4 = (B_ * S_ * I_) / 4;
    cvt_x_bf16<<<(n4 + 255) / 256, 256, 0, stream>>>((const F4*)x, (US4*)xbf, n4);
  }
  {
    int n4 = (3 * H_ * KC_) / 4;
    build_wcat<<<(n4 + 255) / 256, 256, 0, stream>>>(whh, wih, wcat, n4);
  }
  init_hdrop<<<(B_ * H_ + 255) / 256, 256, 0, stream>>>(h0, dmask, mask, hdA);

  for (int s = 0; s < S_; ++s) {
    const unsigned short* hc = (s & 1) ? hdB : hdA;
    unsigned short*       hn = (s & 1) ? hdA : hdB;
    const float* hprev = (s == 0) ? h0 : (out + (size_t)(s - 1) * H_);
    long hstride = (s == 0) ? (long)H_ : (long)S_ * H_;
    gru_step<<<dim3(H_ / 64), dim3(512), 0, stream>>>(
        xbf, wcat, hc, hn, bih, bhh, dmask, mask, hprev, hstride, out, s);
  }
}
